// Attention_Layer_34565896798210
// MI455X (gfx1250) — matrix operand order repaired and checked by reading, compiled, not run on hardware
//
#include <hip/hip_runtime.h>
#include <math.h>

// CDNA5 / gfx1250: wave32, WMMA f32_16x16x32_f16
typedef __attribute__((ext_vector_type(16))) _Float16 v16h;
typedef __attribute__((ext_vector_type(8)))  _Float16 v8h;
typedef __attribute__((ext_vector_type(8)))  float    v8f;

union FragAB { v16h v; v8h h8[2]; _Float16 e[16]; };
union FragCD { v8f  v; float     e[8];  };

#define NEG_INF_VAL (-1e30f)
#define LN_EPS      (1e-5f)

#define STH 72   // f16 LDS row stride in halves (144B, 16B-aligned, bank-staggered)
#define STF 66   // f32 LDS row stride in floats

// A operand: 16x32 f16 tile from row-major (stride STH) matrix.
// Lane (r,h): row rowbase+r; elements 0..7 <- p[h*8..h*8+7], 8..15 <- p[16+h*8..].
__device__ __forceinline__ v16h load_fragA(const _Float16* M, int rowbase,
                                           int tk, int r, int h) {
    const _Float16* p = M + (rowbase + r) * STH + tk * 32 + h * 8;
    FragAB a;
    a.h8[0] = *(const v8h*)(p);
    a.h8[1] = *(const v8h*)(p + 16);
    return a.v;
}

// B operand: 32x16 f16 tile where the source Mt is stored COLUMN-major:
// Mt[n*STH + k] == B[k][n]. Lane (r,h): n = r, K base = tk*32 + h*8,
// elements 0..7 <- p[0..7], 8..15 <- p[16..23].
__device__ __forceinline__ v16h load_fragB_cm(const _Float16* Mt, int tk, int tn,
                                              int r, int h) {
    const _Float16* p = Mt + (tn * 16 + r) * STH + tk * 32 + h * 8;
    FragAB b;
    b.h8[0] = *(const v8h*)(p);
    b.h8[1] = *(const v8h*)(p + 16);
    return b.v;
}

__global__ __launch_bounds__(128)
void attn_layer_kernel(const float* __restrict__ fi,        // [B,64,64]
                       const float* __restrict__ corr,      // [64,64]
                       const float* __restrict__ ln_gamma,  // [64]
                       const float* __restrict__ ln_beta,   // [64]
                       const float* __restrict__ last_w,    // [1,128]
                       const float* __restrict__ last_b,    // [1]
                       float* __restrict__ out)             // [B,64,1]
{
    __shared__ __align__(16) _Float16 sFi [64 * STH]; // fi  row-major  (A ph1; B of fi^T ph2)
    __shared__ __align__(16) _Float16 sFiT[64 * STH]; // fi  col-major  (B ph4)
    __shared__ __align__(16) _Float16 sCt [64 * STH]; // C   col-major  (B ph1)
    __shared__ __align__(16) _Float16 sX  [64 * STH]; // fiC then alpha (A ph2/ph4)
    __shared__ __align__(16) float    sF  [64 * STF]; // beta then vi (f32)
    __shared__ __align__(16) float    sFiF[64 * STF]; // fi f32 (final concat dot)

    const int b    = blockIdx.x;
    const int tid  = threadIdx.x;     // 0..127
    const int lane = tid & 31;
    const int r    = lane & 15;
    const int h    = lane >> 4;
    const int rowbase = (tid >> 5) * 16;   // wave's 16-row strip

    const float* fib = fi + (size_t)b * 64 * 64;

    // ---- Stage fi & C into LDS: row-major f16/f32 + transposed f16 copies ----
    for (int idx = tid; idx < 64 * 64; idx += 128) {
        int row = idx >> 6, col = idx & 63;
        float vf = fib[idx];
        float vc = corr[idx];
        _Float16 hf = (_Float16)vf;
        sFiF[row * STF + col] = vf;
        sFi [row * STH + col] = hf;
        sFiT[col * STH + row] = hf;            // fi^T: [d][j]
        sCt [col * STH + row] = (_Float16)vc;  // C^T:  [e][d]
    }
    __syncthreads();

    v16h afrag[4];

    // ---- Phase 1: fiC = fi @ C ; store f16 into sX ----
#pragma unroll
    for (int tk = 0; tk < 4; ++tk) afrag[tk] = load_fragA(sFi, rowbase, tk, r, h);
#pragma unroll
    for (int tn = 0; tn < 4; ++tn) {
        v8f acc = {};
#pragma unroll
        for (int tk = 0; tk < 4; ++tk) {
            v16h bb = load_fragB_cm(sCt, tk, tn, r, h);
            acc = __builtin_amdgcn_wmma_f32_16x16x32_f16(
                false, afrag[tk], false, bb, (short)0, acc, false, false);
        }
        FragCD d; d.v = acc;
#pragma unroll
        for (int v = 0; v < 8; ++v)
            sX[(rowbase + h * 8 + v) * STH + tn * 16 + r] = (_Float16)d.e[v];
    }
    __syncthreads();

    // ---- Phase 2: beta = fiC @ fi^T ; diag -> -inf ; store f32 in sF ----
    // B[k][n] = fi[n][k]  => column-major view of B is exactly row-major sFi.
#pragma unroll
    for (int tk = 0; tk < 4; ++tk) afrag[tk] = load_fragA(sX, rowbase, tk, r, h);
#pragma unroll
    for (int tn = 0; tn < 4; ++tn) {
        v8f acc = {};
#pragma unroll
        for (int tk = 0; tk < 4; ++tk) {
            v16h bb = load_fragB_cm(sFi, tk, tn, r, h);
            acc = __builtin_amdgcn_wmma_f32_16x16x32_f16(
                false, afrag[tk], false, bb, (short)0, acc, false, false);
        }
        FragCD d; d.v = acc;
#pragma unroll
        for (int v = 0; v < 8; ++v) {
            int row = rowbase + h * 8 + v;
            int col = tn * 16 + r;
            sF[row * STF + col] = (row == col) ? NEG_INF_VAL : d.e[v];
        }
    }
    __syncthreads();

    // ---- Phase 3: row softmax ; alpha (f16) into sX (fiC dead) ----
    if (tid < 64) {
        float* row = sF + tid * STF;
        float m = NEG_INF_VAL;
#pragma unroll 8
        for (int j = 0; j < 64; ++j) m = fmaxf(m, row[j]);
        float s = 0.f;
#pragma unroll 8
        for (int j = 0; j < 64; ++j) { float e = __expf(row[j] - m); s += e; row[j] = e; }
        float inv = 1.f / s;
#pragma unroll 8
        for (int j = 0; j < 64; ++j)
            sX[tid * STH + j] = (_Float16)(row[j] * inv);
    }
    __syncthreads();

    // ---- Phase 4: vi = alpha @ fi ; store f32 into sF (beta dead) ----
    // B[k][n] = fi[k][n]  => column-major source is sFiT.
#pragma unroll
    for (int tk = 0; tk < 4; ++tk) afrag[tk] = load_fragA(sX, rowbase, tk, r, h);
#pragma unroll
    for (int tn = 0; tn < 4; ++tn) {
        v8f acc = {};
#pragma unroll
        for (int tk = 0; tk < 4; ++tk) {
            v16h bb = load_fragB_cm(sFiT, tk, tn, r, h);
            acc = __builtin_amdgcn_wmma_f32_16x16x32_f16(
                false, afrag[tk], false, bb, (short)0, acc, false, false);
        }
        FragCD d; d.v = acc;
#pragma unroll
        for (int v = 0; v < 8; ++v)
            sF[(rowbase + h * 8 + v) * STF + tn * 16 + r] = d.e[v];
    }
    __syncthreads();

    // ---- Phase 5: layernorm + relu + concat-dot + sigmoid, one row/thread ----
    if (tid < 64) {
        const float* vi = sF + tid * STF;
        float mu = 0.f;
#pragma unroll 8
        for (int d = 0; d < 64; ++d) mu += vi[d];
        mu *= (1.f / 64.f);
        float var = 0.f;
#pragma unroll 8
        for (int d = 0; d < 64; ++d) { float z = vi[d] - mu; var += z * z; }
        var *= (1.f / 64.f);
        float rinv = rsqrtf(var + LN_EPS);

        float acc = last_b[0];
#pragma unroll 8
        for (int d = 0; d < 64; ++d) {
            float vn = (vi[d] - mu) * rinv * ln_gamma[d] + ln_beta[d];
            vn = fmaxf(vn, 0.f);
            acc += sFiF[tid * STF + d] * last_w[d] + vn * last_w[64 + d];
        }
        out[(size_t)b * 64 + tid] = 1.f / (1.f + __expf(-acc));
    }
}

extern "C" void kernel_launch(void* const* d_in, const int* in_sizes, int n_in,
                              void* d_out, int out_size, void* d_ws, size_t ws_size,
                              hipStream_t stream) {
    (void)in_sizes; (void)n_in; (void)d_ws; (void)ws_size; (void)out_size;
    const float* fi       = (const float*)d_in[0];  // [8192,64,64]
    const float* corr     = (const float*)d_in[1];  // [64,64]
    const float* ln_gamma = (const float*)d_in[2];  // [64]
    const float* ln_beta  = (const float*)d_in[3];  // [64]
    const float* last_w   = (const float*)d_in[4];  // [1,128]
    const float* last_b   = (const float*)d_in[5];  // [1]
    float* out = (float*)d_out;                     // [8192,64,1]

    const int B = 8192;
    attn_layer_kernel<<<dim3(B), dim3(128), 0, stream>>>(
        fi, corr, ln_gamma, ln_beta, last_w, last_b, out);
}
